// ParallelMLP_52321291600588
// MI455X (gfx1250) — matrix operand order repaired and checked by reading, not compiled, not run on hardware
//
#include <hip/hip_runtime.h>
#include <hip/hip_bf16.h>
#include <math.h>

// ---------------- problem constants (match reference) ----------------
#define E_     8
#define TOPK_  2
#define H_     1024
#define F_     4096
#define TOKENS_ 4096
#define N_     (TOKENS_ * TOPK_)   // 8192
#define CAP_   1024

typedef __bf16 bf16;
typedef __attribute__((ext_vector_type(8)))  bf16  v8bf;
typedef __attribute__((ext_vector_type(16))) bf16  v16bf;
typedef __attribute__((ext_vector_type(8)))  float v8f;

static __device__ inline v16bf cat8(v8bf lo, v8bf hi) {
  v16bf r;
#pragma unroll
  for (int i = 0; i < 8; ++i) { r[i] = lo[i]; r[i + 8] = hi[i]; }
  return r;
}

static __device__ inline v8bf cvt8(const float* f) {
  v8bf r;
#pragma unroll
  for (int i = 0; i < 8; ++i) r[i] = (bf16)f[i];
  return r;
}

// ---------------------------------------------------------------------
// Kernel 1: routing. One block, 256 threads. Stable counting sort by
// expert (chunk histograms + per-expert exclusive prefix over chunks),
// capacity drop at CAP. Produces:
//   slot_src[E*CAP]  : source token for slot (e,p), -1 => zero row
//   assign_slot[N]   : slot index for assignment i, -1 => dropped
// ---------------------------------------------------------------------
__global__ __launch_bounds__(256) void routing_kernel(
    const int* __restrict__ te, int* __restrict__ slot_src,
    int* __restrict__ assign_slot) {
  __shared__ int hist[E_][256];
  const int t = threadIdx.x;
  const int chunk = N_ / 256;   // 32 assignments per thread, in order
  const int base = t * chunk;

#pragma unroll
  for (int e = 0; e < E_; ++e) hist[e][t] = 0;
  __syncthreads();

  // pass A: per-chunk histogram (column t is private to thread t)
  for (int j = 0; j < chunk; ++j) hist[te[base + j]][t]++;
  __syncthreads();

  // per-expert exclusive prefix across the 256 chunks (stable order)
  if (t < E_) {
    int run = 0;
    for (int c = 0; c < 256; ++c) { int v = hist[t][c]; hist[t][c] = run; run += v; }
  }
  __syncthreads();

  // pass C: assign within-bin positions in original (stable) order
  for (int j = 0; j < chunk; ++j) {
    int i  = base + j;
    int e  = te[i];
    int p  = hist[e][t]++;            // chunkBase + local running count
    if (p < CAP_) {
      int slot = e * CAP_ + p;
      slot_src[slot]  = i >> 1;       // token = i / TOPK
      assign_slot[i]  = slot;
    } else {
      assign_slot[i]  = -1;
    }
  }
  __syncthreads();

  // fill unoccupied slots with -1 (zero-padded rows)
  for (int e = 0; e < E_; ++e) {
    int tot = hist[e][255];           // total tokens routed to expert e
    if (tot > CAP_) tot = CAP_;
    for (int p = tot + t; p < CAP_; p += 256) slot_src[e * CAP_ + p] = -1;
  }
}

// ---------------------------------------------------------------------
// GEMM1: h = gelu(buf @ w1)  per expert.
//   A: gathered token rows (fp32 x -> bf16), via slot_src  [CAP x H]
//   B: w1[e] (fp32 -> bf16 on the fly)                     [H  x F]
//   out: hbuf bf16                                         [E*CAP x F]
// block = 256 threads = 8 wave32; tile = 128x128; K-slab = 32.
// ---------------------------------------------------------------------
__global__ __launch_bounds__(256) void gemm1_kernel(
    const float* __restrict__ x, const float* __restrict__ w1,
    const int* __restrict__ slot_src, bf16* __restrict__ hbuf) {
  const int e  = blockIdx.z;
  const int n0 = blockIdx.x * 128;           // F tile
  const int m0 = blockIdx.y * 128;           // CAP tile
  const float* __restrict__ w = w1 + (size_t)e * H_ * F_;

  __shared__ bf16 As[128][40];               // [row][k]  (pitch 40 = 80B)
  __shared__ bf16 Bs[128][40];               // [col][k]

  const int t    = threadIdx.x;
  const int r    = t & 127;                  // row / col handled by this loader
  const int kh   = t >> 7;                   // which 16-wide K half
  const int wave = t >> 5;
  const int lane = t & 31;
  const int wm   = (wave >> 2) * 64;         // wave row base within tile
  const int wn   = (wave & 3) * 32;          // wave col base within tile
  const int lm   = lane & 15;
  const int lq   = lane >> 4;

  const int tok = slot_src[e * CAP_ + m0 + r];

  v8f acc[4][2];
#pragma unroll
  for (int mi = 0; mi < 4; ++mi)
#pragma unroll
    for (int ni = 0; ni < 2; ++ni) acc[mi][ni] = 0.0f;

  for (int k0 = 0; k0 < H_; k0 += 32) {
    // ---- stage A (gather + convert): row m0+r, K = k0 + kh*16 + [0,16)
    {
      float a[16];
      if (tok >= 0) {
        const float4* p = (const float4*)(x + (size_t)tok * H_ + k0 + kh * 16);
        float4 f0 = p[0], f1 = p[1], f2 = p[2], f3 = p[3];
        a[0]=f0.x; a[1]=f0.y; a[2]=f0.z; a[3]=f0.w;
        a[4]=f1.x; a[5]=f1.y; a[6]=f1.z; a[7]=f1.w;
        a[8]=f2.x; a[9]=f2.y; a[10]=f2.z; a[11]=f2.w;
        a[12]=f3.x; a[13]=f3.y; a[14]=f3.z; a[15]=f3.w;
      } else {
#pragma unroll
        for (int i = 0; i < 16; ++i) a[i] = 0.0f;
      }
      *(v8bf*)&As[r][kh * 16]     = cvt8(a);
      *(v8bf*)&As[r][kh * 16 + 8] = cvt8(a + 8);
    }
    // ---- stage B (convert + transpose to [col][k]): col n0+r
    {
      float b[16];
#pragma unroll
      for (int kk = 0; kk < 16; ++kk)
        b[kk] = w[(size_t)(k0 + kh * 16 + kk) * F_ + n0 + r];
      *(v8bf*)&Bs[r][kh * 16]     = cvt8(b);
      *(v8bf*)&Bs[r][kh * 16 + 8] = cvt8(b + 8);
    }
    if (k0 + 32 < H_)
      __builtin_prefetch(&w[(size_t)(k0 + 32 + kh * 16) * F_ + n0 + r], 0, 1);
    __syncthreads();

    // ---- fragments in contiguous k order (k = 16*half + s), same for A and B
    // A 16x32: lanes 0-15 row M, K0..15; lanes 16-31 row M, K16..31
    v16bf afr[4];
#pragma unroll
    for (int mi = 0; mi < 4; ++mi) {
      int row = wm + mi * 16 + lm;
      v8bf lo = *(const v8bf*)&As[row][lq * 16];
      v8bf hi = *(const v8bf*)&As[row][lq * 16 + 8];
      afr[mi] = cat8(lo, hi);
    }
    // B 32x16: lanes 0-15 col N, K0..15; lanes 16-31 col N, K16..31
    v16bf bfr[2];
#pragma unroll
    for (int ni = 0; ni < 2; ++ni) {
      int col = wn + ni * 16 + lm;
      v8bf lo = *(const v8bf*)&Bs[col][lq * 16];
      v8bf hi = *(const v8bf*)&Bs[col][lq * 16 + 8];
      bfr[ni] = cat8(lo, hi);
    }
#pragma unroll
    for (int mi = 0; mi < 4; ++mi)
#pragma unroll
      for (int ni = 0; ni < 2; ++ni)
        acc[mi][ni] = __builtin_amdgcn_wmma_f32_16x16x32_bf16(
            false, afr[mi], false, bfr[ni], (short)0, acc[mi][ni], false, false);
    __syncthreads();
  }

  // ---- epilogue: exact-erf GELU, store bf16
#pragma unroll
  for (int mi = 0; mi < 4; ++mi)
#pragma unroll
    for (int ni = 0; ni < 2; ++ni) {
      int col = n0 + wn + ni * 16 + lm;
#pragma unroll
      for (int v = 0; v < 8; ++v) {
        int row  = m0 + wm + mi * 16 + lq * 8 + v;
        float xv = acc[mi][ni][v];
        float g  = 0.5f * xv * (1.0f + erff(xv * 0.70710678118654752f));
        hbuf[((size_t)(e * CAP_) + row) * F_ + col] = (bf16)g;
      }
    }
}

// ---------------------------------------------------------------------
// GEMM2: y = h @ w2 per expert.  A: hbuf bf16 [CAP x F]; B: w2[e] fp32->bf16
// [F x H]; out: ybuf fp32 [E*CAP x H]. Same tiling as GEMM1.
// A tile is staged with CDNA5 async global->LDS copies (ASYNCcnt path):
// bf16 data needs no conversion, so it never touches VGPRs.
// ---------------------------------------------------------------------
__global__ __launch_bounds__(256) void gemm2_kernel(
    const bf16* __restrict__ hbuf, const float* __restrict__ w2,
    float* __restrict__ ybuf) {
  const int e  = blockIdx.z;
  const int n0 = blockIdx.x * 128;           // H tile
  const int m0 = blockIdx.y * 128;           // CAP tile
  const float* __restrict__ w = w2 + (size_t)e * F_ * H_;

  __shared__ bf16 As[128][40];
  __shared__ bf16 Bs[128][40];

  const int t    = threadIdx.x;
  const int r    = t & 127;
  const int kh   = t >> 7;
  const int wave = t >> 5;
  const int lane = t & 31;
  const int wm   = (wave >> 2) * 64;
  const int wn   = (wave & 3) * 32;
  const int lm   = lane & 15;
  const int lq   = lane >> 4;

  // per-thread LDS destination byte offset (wave-relative; low 32 bits of
  // the flat shared-memory address ARE the LDS offset per ISA 10.2)
  const unsigned lds_dst = (unsigned)(uintptr_t)&As[r][kh * 16];
  const bf16* agbase = hbuf + ((size_t)(e * CAP_) + m0 + r) * F_ + kh * 16;

  v8f acc[4][2];
#pragma unroll
  for (int mi = 0; mi < 4; ++mi)
#pragma unroll
    for (int ni = 0; ni < 2; ++ni) acc[mi][ni] = 0.0f;

  for (int k0 = 0; k0 < F_; k0 += 32) {
    // ---- stage A: async global->LDS, 32B per thread, no VGPR staging.
    // Same INST_OFFSET applies to both global and LDS sides; the two 16B
    // chunks are contiguous on both sides.
    {
      const bf16* gptr = agbase + k0;
      asm volatile(
          "global_load_async_to_lds_b128 %0, %1, off\n\t"
          "global_load_async_to_lds_b128 %0, %1, off offset:16"
          :
          : "v"(lds_dst), "v"(gptr)
          : "memory");
    }
    // ---- stage B: fp32 strided -> bf16, transposed to [col][k]
    {
      float b[16];
#pragma unroll
      for (int kk = 0; kk < 16; ++kk)
        b[kk] = w[(size_t)(k0 + kh * 16 + kk) * H_ + n0 + r];
      *(v8bf*)&Bs[r][kh * 16]     = cvt8(b);
      *(v8bf*)&Bs[r][kh * 16 + 8] = cvt8(b + 8);
    }
    if (k0 + 32 < F_)
      __builtin_prefetch(&w[(size_t)(k0 + 32 + kh * 16) * H_ + n0 + r], 0, 1);
    // complete this wave's async copies before the workgroup barrier
    asm volatile("s_wait_asynccnt 0x0" ::: "memory");
    __syncthreads();

    v16bf afr[4];
#pragma unroll
    for (int mi = 0; mi < 4; ++mi) {
      int row = wm + mi * 16 + lm;
      v8bf lo = *(const v8bf*)&As[row][lq * 16];
      v8bf hi = *(const v8bf*)&As[row][lq * 16 + 8];
      afr[mi] = cat8(lo, hi);
    }
    v16bf bfr[2];
#pragma unroll
    for (int ni = 0; ni < 2; ++ni) {
      int col = wn + ni * 16 + lm;
      v8bf lo = *(const v8bf*)&Bs[col][lq * 16];
      v8bf hi = *(const v8bf*)&Bs[col][lq * 16 + 8];
      bfr[ni] = cat8(lo, hi);
    }
#pragma unroll
    for (int mi = 0; mi < 4; ++mi)
#pragma unroll
      for (int ni = 0; ni < 2; ++ni)
        acc[mi][ni] = __builtin_amdgcn_wmma_f32_16x16x32_bf16(
            false, afr[mi], false, bfr[ni], (short)0, acc[mi][ni], false, false);
    __syncthreads();
  }

#pragma unroll
  for (int mi = 0; mi < 4; ++mi)
#pragma unroll
    for (int ni = 0; ni < 2; ++ni) {
      int col = n0 + wn + ni * 16 + lm;
#pragma unroll
      for (int v = 0; v < 8; ++v) {
        int row = m0 + wm + mi * 16 + lq * 8 + v;
        ybuf[((size_t)(e * CAP_) + row) * H_ + col] = acc[mi][ni][v];
      }
    }
}

// ---------------------------------------------------------------------
// Kernel 4: token-side weighted un-permute + bias (no atomics).
// out[tok] = bias + sum_{j in {2tok, 2tok+1}} ew[j] * y[assign_slot[j]]
// ---------------------------------------------------------------------
__global__ __launch_bounds__(256) void scatter_kernel(
    const float* __restrict__ ybuf, const int* __restrict__ assign_slot,
    const float* __restrict__ ew, const float* __restrict__ bias,
    float* __restrict__ out) {
  const int tk = blockIdx.x;
  const int h  = threadIdx.x * 4;            // 256 threads * 4 = H exactly
  const int j0 = assign_slot[2 * tk];
  const int j1 = assign_slot[2 * tk + 1];
  const float g0 = ew[2 * tk];
  const float g1 = ew[2 * tk + 1];

  float4 r = *(const float4*)(bias + h);
  if (j0 >= 0) {
    float4 y = *(const float4*)(ybuf + (size_t)j0 * H_ + h);
    r.x += g0 * y.x; r.y += g0 * y.y; r.z += g0 * y.z; r.w += g0 * y.w;
  }
  if (j1 >= 0) {
    float4 y = *(const float4*)(ybuf + (size_t)j1 * H_ + h);
    r.x += g1 * y.x; r.y += g1 * y.y; r.z += g1 * y.z; r.w += g1 * y.w;
  }
  *(float4*)(out + (size_t)tk * H_ + h) = r;
}

// ---------------------------------------------------------------------
extern "C" void kernel_launch(void* const* d_in, const int* in_sizes, int n_in,
                              void* d_out, int out_size, void* d_ws, size_t ws_size,
                              hipStream_t stream) {
  const float* x    = (const float*)d_in[0];
  // d_in[1] = scores : unused by the reference computation
  const float* ew   = (const float*)d_in[2];
  const int*   te   = (const int*)d_in[3];
  const float* w1   = (const float*)d_in[4];
  const float* w2   = (const float*)d_in[5];
  const float* bias = (const float*)d_in[6];
  float* out        = (float*)d_out;

  // workspace layout (~96.1 MB total)
  char* ws          = (char*)d_ws;
  int*  slot_src    = (int*)ws;                                   // E*CAP ints
  int*  assign_slot = (int*)(ws + 32768);                         // N ints
  bf16* hbuf        = (bf16*)(ws + 65536);                        // E*CAP*F bf16 (64MB)
  float* ybuf       = (float*)(ws + 65536 + (size_t)E_ * CAP_ * F_ * 2); // 32MB

  routing_kernel<<<1, 256, 0, stream>>>(te, slot_src, assign_slot);
  gemm1_kernel<<<dim3(F_ / 128, CAP_ / 128, E_), 256, 0, stream>>>(x, w1, slot_src, hbuf);
  gemm2_kernel<<<dim3(H_ / 128, CAP_ / 128, E_), 256, 0, stream>>>(hbuf, w2, ybuf);
  scatter_kernel<<<TOKENS_, 256, 0, stream>>>(ybuf, assign_slot, ew, bias, out);
}
